// SinusoidalPositionalAttention_79654463471797
// MI455X (gfx1250) — matrix operand order repaired and checked by reading, not compiled, not run on hardware
//
#include <hip/hip_runtime.h>
#include <hip/hip_bf16.h>
#include <math.h>

#define N_  4
#define L_  768
#define H_  8
#define E_  64
#define DP_ 192   // augmented head dim = 3*E

typedef __attribute__((ext_vector_type(16))) _Float16 v16h;
typedef __attribute__((ext_vector_type(8)))  _Float16 v8h;
typedef __attribute__((ext_vector_type(8)))  float    v8f;

__device__ __forceinline__ float sigmoidf_(float x) { return 1.0f / (1.0f + __expf(-x)); }
__device__ __forceinline__ float softplusf_(float x) { return (x > 20.0f) ? x : log1pf(__expf(x)); }

// ---------------------------------------------------------------------------
// Prep: build augmented Q'/K' (f16, [N,H,L,192]) and transposed V (f16,
// [N,H,E,S]).  Uses P = (1-gate)*g^2*(cos(phq)cos(phk)+sin(phq)sin(phk))
// + gate, folding softmax_temp = 1/sqrt(E) = 0.125 into Q'.
// ---------------------------------------------------------------------------
__global__ __launch_bounds__(256) void spa_prep(
    const float* __restrict__ q, const float* __restrict__ k, const float* __restrict__ v,
    const float* __restrict__ freqs, const float* __restrict__ offs,
    const float* __restrict__ gains, const float* __restrict__ gate,
    _Float16* __restrict__ Qp, _Float16* __restrict__ Kp, _Float16* __restrict__ Vt)
{
    int idx = blockIdx.x * blockDim.x + threadIdx.x;
    if (idx >= N_ * L_ * H_ * E_) return;
    // idx == ((n*L + l)*H + h)*E + d  (matches (N,L,H,E) input layout)
    int d = idx & (E_ - 1);
    int h = (idx >> 6) & (H_ - 1);
    int l = (idx >> 9) % L_;
    int n = idx / (E_ * H_ * L_);
    int hd = h * E_ + d;

    const float PI2 = 6.28318530717958647692f;
    float w    = PI2 * 0.5f * sigmoidf_(freqs[hd]);   // 2*pi*f, K=1
    float off  = offs[hd];
    float g    = softplusf_(gains[hd]);
    float gt   = gate[hd];
    float coef = (1.0f - gt) * g * g;
    float ph   = w * (float)l;

    float sq, cq, sk, ck;
    __sincosf(ph + off, &sq, &cq);
    __sincosf(ph, &sk, &ck);

    float qv = q[idx], kv = k[idx], vv = v[idx];
    const float temp = 0.125f;   // 1/sqrt(64)

    size_t ob = ((size_t)((n * H_ + h) * L_ + l)) * DP_;
    Qp[ob + d]          = (_Float16)(temp * coef * cq * qv);
    Qp[ob + E_ + d]     = (_Float16)(temp * coef * sq * qv);
    Qp[ob + 2 * E_ + d] = (_Float16)(temp * gt * qv);
    Kp[ob + d]          = (_Float16)(ck * kv);
    Kp[ob + E_ + d]     = (_Float16)(sk * kv);
    Kp[ob + 2 * E_ + d] = (_Float16)kv;
    // transposed V: [N,H,E,S] so PV B-fragments are contiguous along s
    Vt[((size_t)((n * H_ + h) * E_ + d)) * L_ + l] = (_Float16)vv;
}

// ---------------------------------------------------------------------------
// Flash attention: one wave32 per (n, h, 16-row l-tile).
// Per 32-col s-chunk: issue all 16 B-fragment loads up front (12 K + 4 V,
// each its own registers so the scheduler can keep them in flight), then
// 12 QK WMMAs, LDS online softmax, 4 PV WMMAs.
// ---------------------------------------------------------------------------
__global__ __launch_bounds__(32) void spa_attn(
    const _Float16* __restrict__ Qp, const _Float16* __restrict__ Kp,
    const _Float16* __restrict__ Vt,
    const float* __restrict__ mask, const float* __restrict__ klen,
    float* __restrict__ out)
{
    __shared__ __align__(16) float    lds_qk[16 * 32];
    __shared__ __align__(16) _Float16 lds_p[16 * 32];
    __shared__ float lds_alpha[16];
    __shared__ float lds_m[16];
    __shared__ float lds_s[16];

    const int lane = threadIdx.x;      // 0..31
    const int m16  = lane & 15;
    const int half = lane >> 4;
    const int l0   = blockIdx.x * 16;
    const int h    = blockIdx.y;
    const int n    = blockIdx.z;

    const _Float16* Qb = Qp + ((size_t)(n * H_ + h) * L_) * DP_;
    const _Float16* Kb = Kp + ((size_t)(n * H_ + h) * L_) * DP_;
    const _Float16* Vb = Vt + ((size_t)(n * H_ + h) * E_) * L_;   // [E][S]

    // Resident Q A-fragments: 6 fragments of 16x32 halves over d'=192.
    // A layout: lane=(m,half), element j -> k = 16*half + j
    v16h qa[6];
#pragma unroll
    for (int c = 0; c < 6; ++c) {
        const _Float16* rp = Qb + (size_t)(l0 + m16) * DP_ + c * 32 + half * 16;
        v8h lo = *(const v8h*)(rp);        // k-local 16*half + 0..7
        v8h hi = *(const v8h*)(rp + 8);    // k-local 16*half + 8..15
#pragma unroll
        for (int j = 0; j < 8; ++j) { qa[c][j] = lo[j]; qa[c][8 + j] = hi[j]; }
    }

    v8f o[4] = {v8f{}, v8f{}, v8f{}, v8f{}};
    if (lane < 16) { lds_m[lane] = -3.0e38f; lds_s[lane] = 0.0f; }

    // Per-lane base pointers for the B-fragment loads (advance by 32 cols/iter).
    const _Float16* kp0 = Kb + (size_t)m16 * DP_ + half * 16;          // col group s0
    const _Float16* kp1 = Kb + (size_t)(16 + m16) * DP_ + half * 16;   // col group s0+16
    const _Float16* vp  = Vb + (size_t)m16 * L_ + half * 16;           // V, d-row m16

    for (int s0 = 0; s0 < L_; s0 += 32) {
        // ---- Issue ALL fragment loads for this chunk (independent regs) ----
        v16h kb0[6], kb1[6];
#pragma unroll
        for (int c = 0; c < 6; ++c) {
            kb0[c] = *(const v16h*)(kp0 + (size_t)s0 * DP_ + c * 32);
            kb1[c] = *(const v16h*)(kp1 + (size_t)s0 * DP_ + c * 32);
        }
        v16h vbuf[4];
#pragma unroll
        for (int t = 0; t < 4; ++t)
            vbuf[t] = *(const v16h*)(vp + (size_t)(16 * t) * L_ + s0);

        // --- QK^T: two 16x16 score tiles (cols s0..+15, s0+16..+31) ---
        v8f d0 = {}, d1 = {};
#pragma unroll
        for (int c = 0; c < 6; ++c) {
            d0 = __builtin_amdgcn_wmma_f32_16x16x32_f16(false, qa[c], false, kb0[c],
                                                        (short)0, d0, false, false);
            d1 = __builtin_amdgcn_wmma_f32_16x16x32_f16(false, qa[c], false, kb1[c],
                                                        (short)0, d1, false, false);
        }

        // Spill scores to LDS: D layout lane=(col,half), reg r -> row r+8*half
#pragma unroll
        for (int r = 0; r < 8; ++r) {
            lds_qk[(r + 8 * half) * 32 + m16]      = d0[r];
            lds_qk[(r + 8 * half) * 32 + 16 + m16] = d1[r];
        }

        // --- Online softmax: lane m (0..15) owns row m of the tile ---
        if (lane < 16) {
            const int lrow = l0 + lane;
            float mold = lds_m[lane];
            float vals[32];
            float rowmax = mold;
#pragma unroll
            for (int j = 0; j < 32; ++j) {
                float t = lds_qk[lane * 32 + j]
                        + 0.125f * (mask[(size_t)lrow * L_ + s0 + j]
                                  + klen[(size_t)n * L_ + s0 + j]);
                vals[j] = t;
                rowmax = fmaxf(rowmax, t);
            }
            float alpha = __expf(mold - rowmax);
            float s = lds_s[lane] * alpha;
#pragma unroll
            for (int j = 0; j < 32; ++j) {
                float p = __expf(vals[j] - rowmax);
                s += p;
                lds_p[lane * 32 + j] = (_Float16)p;   // row-major [16][32]
            }
            lds_m[lane] = rowmax;
            lds_s[lane] = s;
            lds_alpha[lane] = alpha;
        }

        // Rescale O accumulators by per-row alpha.
        float al[8];
#pragma unroll
        for (int r = 0; r < 8; ++r) al[r] = lds_alpha[r + 8 * half];
#pragma unroll
        for (int t = 0; t < 4; ++t)
#pragma unroll
            for (int r = 0; r < 8; ++r) o[t][r] *= al[r];

        // P fragment (A layout) from LDS.
        v16h pa;
        {
            v8h lo = *(const v8h*)&lds_p[m16 * 32 + 16 * half];
            v8h hi = *(const v8h*)&lds_p[m16 * 32 + 16 * half + 8];
#pragma unroll
            for (int j = 0; j < 8; ++j) { pa[j] = lo[j]; pa[8 + j] = hi[j]; }
        }

        // PV: V fragments were prefetched at the top of the iteration.
#pragma unroll
        for (int t = 0; t < 4; ++t) {
            o[t] = __builtin_amdgcn_wmma_f32_16x16x32_f16(false, pa, false, vbuf[t],
                                                          (short)0, o[t], false, false);
        }
    }

    // Normalize and store: out is (N,L,H,E) f32.
    float inv[8];
#pragma unroll
    for (int r = 0; r < 8; ++r) inv[r] = 1.0f / lds_s[r + 8 * half];
#pragma unroll
    for (int r = 0; r < 8; ++r) {
        int lrow = l0 + r + 8 * half;
        size_t ob = ((size_t)(n * L_ + lrow) * H_ + h) * E_;
        out[ob + m16]      = o[0][r] * inv[r];
        out[ob + 16 + m16] = o[1][r] * inv[r];
        out[ob + 32 + m16] = o[2][r] * inv[r];
        out[ob + 48 + m16] = o[3][r] * inv[r];
    }
}

// ---------------------------------------------------------------------------
extern "C" void kernel_launch(void* const* d_in, const int* in_sizes, int n_in,
                              void* d_out, int out_size, void* d_ws, size_t ws_size,
                              hipStream_t stream) {
    const float* q     = (const float*)d_in[0];
    const float* k     = (const float*)d_in[1];
    const float* v     = (const float*)d_in[2];
    const float* mask  = (const float*)d_in[3];
    const float* klen  = (const float*)d_in[4];
    const float* freqs = (const float*)d_in[5];
    const float* offs  = (const float*)d_in[6];
    const float* gains = (const float*)d_in[7];
    const float* gate  = (const float*)d_in[8];
    float* out = (float*)d_out;

    _Float16* Qp = (_Float16*)d_ws;                      // [N,H,L,192] f16
    _Float16* Kp = Qp + (size_t)N_ * H_ * L_ * DP_;      // [N,H,L,192] f16
    _Float16* Vt = Kp + (size_t)N_ * H_ * L_ * DP_;      // [N,H,E,S]   f16

    int total = N_ * L_ * H_ * E_;
    spa_prep<<<(total + 255) / 256, 256, 0, stream>>>(q, k, v, freqs, offs, gains, gate,
                                                      Qp, Kp, Vt);
    dim3 grid(L_ / 16, H_, N_);
    spa_attn<<<grid, 32, 0, stream>>>(Qp, Kp, Vt, mask, klen, out);
}
